// QuantMeissonicSingleTransformerBlock_64269890617873
// MI455X (gfx1250) — 16-bit and 8-bit matrix operand order repaired and checked by reading, compiled, not run on hardware
//
#include <hip/hip_runtime.h>


typedef __attribute__((ext_vector_type(16))) _Float16 v16h;
typedef __attribute__((ext_vector_type(8)))  _Float16 v8h;
typedef __attribute__((ext_vector_type(16))) __bf16   v16bf;
typedef __attribute__((ext_vector_type(8)))  __bf16   v8bf;
typedef __attribute__((ext_vector_type(8)))  float    v8f;
typedef __attribute__((ext_vector_type(8)))  int      v8i;

#define DIMC   2048
#define SEQ    1024
#define BATCH  2
#define HEADS  16
#define HD     128
#define MLPD   8192
#define CATD   10240
#define MTOK   2048
#define NQKV   6144

static constexpr size_t MB1 = 1ull << 20;
static constexpr size_t OFF_EMB    = 0;
static constexpr size_t OFF_XSCALE = 64  * 1024;
static constexpr size_t OFF_W8S    = 128 * 1024;
static constexpr size_t OFF_WMLPS  = 192 * 1024;
static constexpr size_t OFF_WOUTS  = 256 * 1024;
static constexpr size_t OFF_NX     = 1   * MB1;
static constexpr size_t OFF_XQ     = 9   * MB1;
static constexpr size_t OFF_W8     = 13  * MB1;
static constexpr size_t OFF_WMLP   = 25  * MB1;
static constexpr size_t OFF_WOUT   = 57  * MB1;
static constexpr size_t OFF_QKV    = 97  * MB1;
static constexpr size_t OFF_QH     = 145 * MB1;
static constexpr size_t OFF_KH     = 153 * MB1;
static constexpr size_t OFF_VH     = 161 * MB1;
static constexpr size_t OFF_ATTN   = 169 * MB1;
static constexpr size_t OFF_CAT    = 185 * MB1;

__global__ __launch_bounds__(256) void k_wq8(const float* __restrict__ wq,
                                             const float* __restrict__ wk,
                                             const float* __restrict__ wv,
                                             signed char* __restrict__ w8,
                                             float* __restrict__ w8s) {
  __shared__ float red[256];
  const int r = blockIdx.x, tid = threadIdx.x;
  const float* src = (r < 2048) ? wq + (size_t)r * DIMC
                   : (r < 4096) ? wk + (size_t)(r - 2048) * DIMC
                                : wv + (size_t)(r - 4096) * DIMC;
  float amax = 0.f;
  for (int k = tid; k < DIMC; k += 256) amax = fmaxf(amax, fabsf(src[k]));
  red[tid] = amax; __syncthreads();
  for (int st = 128; st; st >>= 1) { if (tid < st) red[tid] = fmaxf(red[tid], red[tid + st]); __syncthreads(); }
  const float s = fmaxf(red[0] / 127.f, 1e-8f);
  if (tid == 0) w8s[r] = s;
  for (int k = tid; k < DIMC; k += 256)
    w8[(size_t)r * DIMC + k] = (signed char)fminf(fmaxf(rintf(src[k] / s), -127.f), 127.f);
}

__global__ __launch_bounds__(256) void k_wqbf(const float* __restrict__ src,
                                              __bf16* __restrict__ dst,
                                              float* __restrict__ scales, int cols) {
  __shared__ float red[256];
  const int r = blockIdx.x, tid = threadIdx.x;
  const float* wr = src + (size_t)r * cols;
  float amax = 0.f;
  for (int k = tid; k < cols; k += 256) amax = fmaxf(amax, fabsf(wr[k]));
  red[tid] = amax; __syncthreads();
  for (int st = 128; st; st >>= 1) { if (tid < st) red[tid] = fmaxf(red[tid], red[tid + st]); __syncthreads(); }
  const float s = fmaxf(red[0] / 127.f, 1e-8f);
  if (tid == 0) scales[r] = s;
  for (int k = tid; k < cols; k += 256)
    dst[(size_t)r * cols + k] = (__bf16)fminf(fmaxf(rintf(wr[k] / s), -127.f), 127.f);
}

__global__ __launch_bounds__(256) void k_adaln(const float* __restrict__ temb,
                                               const float* __restrict__ w,
                                               const float* __restrict__ bias,
                                               float* __restrict__ emb) {
  __shared__ float red[256];
  const int o = blockIdx.x, tid = threadIdx.x;
  const float* wr = w + (size_t)o * DIMC;
  float amax = 0.f;
  for (int k = tid; k < DIMC; k += 256) amax = fmaxf(amax, fabsf(wr[k]));
  red[tid] = amax; __syncthreads();
  for (int st = 128; st; st >>= 1) { if (tid < st) red[tid] = fmaxf(red[tid], red[tid + st]); __syncthreads(); }
  const float s = fmaxf(red[0] / 127.f, 1e-8f);
  __syncthreads();
  float a0 = 0.f, a1 = 0.f;
  for (int k = tid; k < DIMC; k += 256) {
    const float qw = fminf(fmaxf(rintf(wr[k] / s), -127.f), 127.f) * s;
    float t0 = temb[k];        t0 = t0 / (1.f + __expf(-t0));
    float t1 = temb[DIMC + k]; t1 = t1 / (1.f + __expf(-t1));
    a0 += qw * t0; a1 += qw * t1;
  }
  red[tid] = a0; __syncthreads();
  for (int st = 128; st; st >>= 1) { if (tid < st) red[tid] += red[tid + st]; __syncthreads(); }
  if (tid == 0) emb[o] = red[0] + bias[o];
  __syncthreads();
  red[tid] = a1; __syncthreads();
  for (int st = 128; st; st >>= 1) { if (tid < st) red[tid] += red[tid + st]; __syncthreads(); }
  if (tid == 0) emb[NQKV + o] = red[0] + bias[o];
}

__global__ __launch_bounds__(256) void k_ln(const float* __restrict__ hs,
                                            const float* __restrict__ emb,
                                            __bf16* __restrict__ nx,
                                            signed char* __restrict__ xq,
                                            float* __restrict__ xs) {
  __shared__ float red[256];
  const int m = blockIdx.x, b = m >> 10, tid = threadIdx.x;
  const float* x = hs + (size_t)m * DIMC;
  float v[8];
  float sum = 0.f;
#pragma unroll
  for (int i = 0; i < 8; ++i) { v[i] = x[tid + i * 256]; sum += v[i]; }
  red[tid] = sum; __syncthreads();
  for (int st = 128; st; st >>= 1) { if (tid < st) red[tid] += red[tid + st]; __syncthreads(); }
  const float mu = red[0] / (float)DIMC;
  __syncthreads();
  float ssq = 0.f;
#pragma unroll
  for (int i = 0; i < 8; ++i) { const float d = v[i] - mu; ssq += d * d; }
  red[tid] = ssq; __syncthreads();
  for (int st = 128; st; st >>= 1) { if (tid < st) red[tid] += red[tid + st]; __syncthreads(); }
  const float rstd = rsqrtf(red[0] / (float)DIMC + 1e-6f);
  __syncthreads();
  float amax = 0.f;
#pragma unroll
  for (int i = 0; i < 8; ++i) {
    const int c = tid + i * 256;
    const float sc = emb[(size_t)b * NQKV + DIMC + c];
    const float sh = emb[(size_t)b * NQKV + c];
    const float y = (v[i] - mu) * rstd * (1.f + sc) + sh;
    v[i] = y; amax = fmaxf(amax, fabsf(y));
  }
  red[tid] = amax; __syncthreads();
  for (int st = 128; st; st >>= 1) { if (tid < st) red[tid] = fmaxf(red[tid], red[tid + st]); __syncthreads(); }
  const float s = fmaxf(red[0] / 127.f, 1e-8f);
  if (tid == 0) xs[m] = s;
#pragma unroll
  for (int i = 0; i < 8; ++i) {
    const int c = tid + i * 256;
    nx[(size_t)m * DIMC + c] = (__bf16)v[i];
    xq[(size_t)m * DIMC + c] = (signed char)fminf(fmaxf(rintf(v[i] / s), -127.f), 127.f);
  }
}

__device__ inline v8i load_a8(const signed char* p, int lh) {
  v8i a;
  ((int4*)&a)[0] = *(const int4*)(p +  0 + 16 * lh);
  ((int4*)&a)[1] = *(const int4*)(p + 32 + 16 * lh);
  return a;
}
__device__ inline v8i load_b8(const signed char* p, int lh) {
  v8i b;
  ((int4*)&b)[0] = *(const int4*)(p +  0 + 16 * lh);
  ((int4*)&b)[1] = *(const int4*)(p + 32 + 16 * lh);
  return b;
}
__device__ inline v16bf load_abf(const __bf16* p, int lh) {
  v16bf a;
  *(v8bf*)&a       = *(const v8bf*)(p + 16 * lh);
  *((v8bf*)&a + 1) = *(const v8bf*)(p + 16 * lh + 8);
  return a;
}

__global__ __launch_bounds__(256) void k_gemm_qkv(const signed char* __restrict__ xq,
                                                  const float* __restrict__ xs,
                                                  const signed char* __restrict__ w8,
                                                  const float* __restrict__ ws8,
                                                  const float* __restrict__ bq,
                                                  const float* __restrict__ bk,
                                                  const float* __restrict__ bv,
                                                  float* __restrict__ qkv) {
  const int wave = threadIdx.x >> 5, lane = threadIdx.x & 31;
  const int l16 = lane & 15, lh = lane >> 4;
  const int Mb = blockIdx.x * 128 + (wave & 3) * 32;
  const int Nb = blockIdx.y * 64 + (wave >> 2) * 32;
  v8i c[2][2] = {};
  const signed char* a0p = xq + (size_t)(Mb + l16) * DIMC;
  const signed char* a1p = a0p + (size_t)16 * DIMC;
  const signed char* b0p = w8 + (size_t)(Nb + l16) * DIMC;
  const signed char* b1p = b0p + (size_t)16 * DIMC;
  for (int kc = 0; kc < DIMC; kc += 64) {
    if (kc + 64 < DIMC) { __builtin_prefetch(a0p + kc + 64); __builtin_prefetch(b0p + kc + 64); }
    const v8i a0 = load_a8(a0p + kc, lh);
    const v8i a1 = load_a8(a1p + kc, lh);
    const v8i b0 = load_b8(b0p + kc, lh);
    const v8i b1 = load_b8(b1p + kc, lh);
    c[0][0] = __builtin_amdgcn_wmma_i32_16x16x64_iu8(true, a0, true, b0, c[0][0], false, false);
    c[0][1] = __builtin_amdgcn_wmma_i32_16x16x64_iu8(true, a0, true, b1, c[0][1], false, false);
    c[1][0] = __builtin_amdgcn_wmma_i32_16x16x64_iu8(true, a1, true, b0, c[1][0], false, false);
    c[1][1] = __builtin_amdgcn_wmma_i32_16x16x64_iu8(true, a1, true, b1, c[1][1], false, false);
  }
#pragma unroll
  for (int i = 0; i < 2; ++i)
#pragma unroll
    for (int j = 0; j < 2; ++j) {
      const int n = Nb + j * 16 + l16;
      const float sw = ws8[n];
      const float bias = (n < 2048) ? bq[n] : (n < 4096) ? bk[n - 2048] : bv[n - 4096];
#pragma unroll
      for (int r = 0; r < 8; ++r) {
        const int m = Mb + i * 16 + 8 * lh + r;
        qkv[(size_t)m * NQKV + n] = (float)c[i][j][r] * xs[m] * sw + bias;
      }
    }
}

__global__ __launch_bounds__(256) void k_gemm_mlp(const __bf16* __restrict__ A,
                                                  const __bf16* __restrict__ W,
                                                  const float* __restrict__ wsc,
                                                  const float* __restrict__ bias,
                                                  __bf16* __restrict__ cat) {
  const int wave = threadIdx.x >> 5, lane = threadIdx.x & 31;
  const int l16 = lane & 15, lh = lane >> 4;
  const int Mb = blockIdx.x * 128 + (wave & 3) * 32;
  const int Nb = blockIdx.y * 64 + (wave >> 2) * 32;
  v8f c[2][2] = {};
  const __bf16* a0p = A + (size_t)(Mb + l16) * DIMC;
  const __bf16* a1p = a0p + (size_t)16 * DIMC;
  const __bf16* b0p = W + (size_t)(Nb + l16) * DIMC;
  const __bf16* b1p = b0p + (size_t)16 * DIMC;
  for (int kc = 0; kc < DIMC; kc += 32) {
    if (kc + 32 < DIMC) { __builtin_prefetch(a0p + kc + 32); __builtin_prefetch(b0p + kc + 32); }
    const v16bf a0 = load_abf(a0p + kc, lh);
    const v16bf a1 = load_abf(a1p + kc, lh);
    const v16bf b0 = *(const v16bf*)(b0p + kc + 16 * lh);
    const v16bf b1 = *(const v16bf*)(b1p + kc + 16 * lh);
    c[0][0] = __builtin_amdgcn_wmma_f32_16x16x32_bf16(false, a0, false, b0, (short)0, c[0][0], false, false);
    c[0][1] = __builtin_amdgcn_wmma_f32_16x16x32_bf16(false, a0, false, b1, (short)0, c[0][1], false, false);
    c[1][0] = __builtin_amdgcn_wmma_f32_16x16x32_bf16(false, a1, false, b0, (short)0, c[1][0], false, false);
    c[1][1] = __builtin_amdgcn_wmma_f32_16x16x32_bf16(false, a1, false, b1, (short)0, c[1][1], false, false);
  }
#pragma unroll
  for (int i = 0; i < 2; ++i)
#pragma unroll
    for (int j = 0; j < 2; ++j) {
      const int n = Nb + j * 16 + l16;
      const float sw = wsc[n], bb = bias[n];
#pragma unroll
      for (int r = 0; r < 8; ++r) {
        const int m = Mb + i * 16 + 8 * lh + r;
        const float u = c[i][j][r] * sw + bb;
        const float g = 0.5f * u * (1.f + tanhf(0.7978845608028654f * (u + 0.044715f * u * u * u)));
        cat[(size_t)m * CATD + DIMC + n] = (__bf16)g;
      }
    }
}

__global__ __launch_bounds__(128) void k_qkvhead(const float* __restrict__ qkv,
                                                 const float* __restrict__ rmsq_w,
                                                 const float* __restrict__ rmsk_w,
                                                 const float* __restrict__ cosb,
                                                 const float* __restrict__ sinb,
                                                 _Float16* __restrict__ qh,
                                                 _Float16* __restrict__ kh,
                                                 _Float16* __restrict__ vh) {
  __shared__ float red[128];
  const int mh = blockIdx.x;
  const int m = mh >> 4, h = mh & 15;
  const int b = m >> 10, s = m & 1023;
  const int d = threadIdx.x;
  const float* base = qkv + (size_t)m * NQKV + h * HD;
  float qv = base[d], kv = base[2048 + d], vv = base[4096 + d];

#define RSUM(OUT, X) { red[d] = (X); __syncthreads();                                   \
  for (int st = 64; st; st >>= 1) { if (d < st) red[d] += red[d + st]; __syncthreads(); } \
  OUT = red[0]; __syncthreads(); }
#define RMAX(OUT, X) { red[d] = (X); __syncthreads();                                   \
  for (int st = 64; st; st >>= 1) { if (d < st) red[d] = fmaxf(red[d], red[d + st]); __syncthreads(); } \
  OUT = red[0]; __syncthreads(); }

  float msq, msk;
  RSUM(msq, qv * qv); RSUM(msk, kv * kv);
  float qn = qv * rsqrtf(msq / (float)HD + 1e-6f) * rmsq_w[d];
  float kn = kv * rsqrtf(msk / (float)HD + 1e-6f) * rmsk_w[d];
  float aq, ak, av;
  RMAX(aq, fabsf(qn)); RMAX(ak, fabsf(kn)); RMAX(av, fabsf(vv));
  const float sq = fmaxf(aq / 127.f, 1e-8f);
  const float sk = fmaxf(ak / 127.f, 1e-8f);
  const float sv = fmaxf(av / 127.f, 1e-8f);
  qn = fminf(fmaxf(rintf(qn / sq), -127.f), 127.f) * sq;
  kn = fminf(fmaxf(rintf(kn / sk), -127.f), 127.f) * sk;
  const float vq = fminf(fmaxf(rintf(vv / sv), -127.f), 127.f) * sv;
  const float qp = __shfl_xor(qn, 1, 32);
  const float kp = __shfl_xor(kn, 1, 32);
  const float rotq = (d & 1) ? qp : -qp;
  const float rotk = (d & 1) ? kp : -kp;
  const float cs = cosb[(size_t)s * HD + d], sn = sinb[(size_t)s * HD + d];
  const size_t o = ((size_t)((b * HEADS + h) * SEQ + s)) * HD + d;
  qh[o] = (_Float16)(qn * cs + rotq * sn);
  kh[o] = (_Float16)(kn * cs + rotk * sn);
  vh[o] = (_Float16)vq;
#undef RSUM
#undef RMAX
}

__global__ __launch_bounds__(256) void k_attn(const _Float16* __restrict__ qh,
                                              const _Float16* __restrict__ kh,
                                              const _Float16* __restrict__ vh,
                                              float* __restrict__ attn) {
  __shared__ __align__(64) _Float16 VT[HD * 32];
  __shared__ __align__(64) _Float16 Pt[8][16 * 32];
  const int bh = blockIdx.x;
  const int b = bh >> 4, h = bh & 15;
  const int wave = threadIdx.x >> 5, lane = threadIdx.x & 31;
  const int l16 = lane & 15, lh = lane >> 4;
  const _Float16* qp = qh + (size_t)bh * SEQ * HD;
  const _Float16* kp = kh + (size_t)bh * SEQ * HD;
  const _Float16* vp = vh + (size_t)bh * SEQ * HD;
  const int qrow = blockIdx.y * 128 + wave * 16 + l16;

  v16h aq[4];
#pragma unroll
  for (int kk = 0; kk < 4; ++kk) {
    const _Float16* r = qp + (size_t)qrow * HD + kk * 32;
    *(v8h*)&aq[kk]       = *(const v8h*)(r + 16 * lh);
    *((v8h*)&aq[kk] + 1) = *(const v8h*)(r + 16 * lh + 8);
  }
  v8f o[8] = {};
  float mrow[8], lrow[8];
#pragma unroll
  for (int r = 0; r < 8; ++r) { mrow[r] = -1e30f; lrow[r] = 0.f; }
  const float scale = 0.08838834764831845f;

  for (int kt = 0; kt < SEQ; kt += 32) {
    __syncthreads();
    {
      const int j = threadIdx.x >> 3;
      const int dbase = (threadIdx.x & 7) * 16;
      const _Float16* vr = vp + (size_t)(kt + j) * HD + dbase;
#pragma unroll
      for (int i = 0; i < 16; ++i) VT[(dbase + i) * 32 + j] = vr[i];
    }
    __syncthreads();

    v8f s0 = {}, s1 = {};
#pragma unroll
    for (int kk = 0; kk < 4; ++kk) {
      const v16h b0 = *(const v16h*)(kp + (size_t)(kt + l16) * HD + kk * 32 + 16 * lh);
      s0 = __builtin_amdgcn_wmma_f32_16x16x32_f16(false, aq[kk], false, b0, (short)0, s0, false, false);
      const v16h b1 = *(const v16h*)(kp + (size_t)(kt + 16 + l16) * HD + kk * 32 + 16 * lh);
      s1 = __builtin_amdgcn_wmma_f32_16x16x32_f16(false, aq[kk], false, b1, (short)0, s1, false, false);
    }
    float fac[8];
#pragma unroll
    for (int r = 0; r < 8; ++r) {
      const float sa = s0[r] * scale, sb = s1[r] * scale;
      float t = fmaxf(sa, sb);
      for (int msk = 1; msk < 16; msk <<= 1) t = fmaxf(t, __shfl_xor(t, msk, 32));
      const float mnew = fmaxf(mrow[r], t);
      const float f = __expf(mrow[r] - mnew);
      const float p0 = __expf(sa - mnew), p1 = __expf(sb - mnew);
      float ps = p0 + p1;
      for (int msk = 1; msk < 16; msk <<= 1) ps += __shfl_xor(ps, msk, 32);
      lrow[r] = lrow[r] * f + ps;
      mrow[r] = mnew; fac[r] = f;
      Pt[wave][(r + 8 * lh) * 32 + l16]      = (_Float16)p0;
      Pt[wave][(r + 8 * lh) * 32 + 16 + l16] = (_Float16)p1;
    }
    v16h ap;
    *(v8h*)&ap       = *(const v8h*)&Pt[wave][l16 * 32 + 16 * lh];
    *((v8h*)&ap + 1) = *(const v8h*)&Pt[wave][l16 * 32 + 16 * lh + 8];
#pragma unroll
    for (int dt = 0; dt < 8; ++dt) {
      v8f oo = o[dt];
#pragma unroll
      for (int r = 0; r < 8; ++r) oo[r] *= fac[r];
      const v16h bv = *(const v16h*)&VT[(dt * 16 + l16) * 32 + 16 * lh];
      o[dt] = __builtin_amdgcn_wmma_f32_16x16x32_f16(false, ap, false, bv, (short)0, oo, false, false);
    }
  }
#pragma unroll
  for (int dt = 0; dt < 8; ++dt)
#pragma unroll
    for (int r = 0; r < 8; ++r) {
      const int row = blockIdx.y * 128 + wave * 16 + r + 8 * lh;
      const int d = dt * 16 + l16;
      attn[((size_t)(b * SEQ + row)) * DIMC + h * HD + d] = o[dt][r] / lrow[r];
    }
}

__global__ __launch_bounds__(256) void k_attnq(const float* __restrict__ attn,
                                               __bf16* __restrict__ cat) {
  __shared__ float red[256];
  const int m = blockIdx.x, tid = threadIdx.x;
  const float* x = attn + (size_t)m * DIMC;
  float amax = 0.f;
#pragma unroll
  for (int i = 0; i < 8; ++i) amax = fmaxf(amax, fabsf(x[tid + i * 256]));
  red[tid] = amax; __syncthreads();
  for (int st = 128; st; st >>= 1) { if (tid < st) red[tid] = fmaxf(red[tid], red[tid + st]); __syncthreads(); }
  const float s = fmaxf(red[0] / 127.f, 1e-8f);
#pragma unroll
  for (int i = 0; i < 8; ++i) {
    const int c = tid + i * 256;
    cat[(size_t)m * CATD + c] = (__bf16)(fminf(fmaxf(rintf(x[c] / s), -127.f), 127.f) * s);
  }
}

__global__ __launch_bounds__(256) void k_gemm_out(const __bf16* __restrict__ A,
                                                  const __bf16* __restrict__ W,
                                                  const float* __restrict__ wsc,
                                                  const float* __restrict__ bias,
                                                  const float* __restrict__ hs,
                                                  const float* __restrict__ emb,
                                                  float* __restrict__ outp) {
  const int wave = threadIdx.x >> 5, lane = threadIdx.x & 31;
  const int l16 = lane & 15, lh = lane >> 4;
  const int Mb = blockIdx.x * 128 + (wave & 3) * 32;
  const int Nb = blockIdx.y * 64 + (wave >> 2) * 32;
  v8f c[2][2] = {};
  const __bf16* a0p = A + (size_t)(Mb + l16) * CATD;
  const __bf16* a1p = a0p + (size_t)16 * CATD;
  const __bf16* b0p = W + (size_t)(Nb + l16) * CATD;
  const __bf16* b1p = b0p + (size_t)16 * CATD;
  for (int kc = 0; kc < CATD; kc += 32) {
    if (kc + 32 < CATD) { __builtin_prefetch(a0p + kc + 32); __builtin_prefetch(b0p + kc + 32); }
    const v16bf a0 = load_abf(a0p + kc, lh);
    const v16bf a1 = load_abf(a1p + kc, lh);
    const v16bf b0 = *(const v16bf*)(b0p + kc + 16 * lh);
    const v16bf b1 = *(const v16bf*)(b1p + kc + 16 * lh);
    c[0][0] = __builtin_amdgcn_wmma_f32_16x16x32_bf16(false, a0, false, b0, (short)0, c[0][0], false, false);
    c[0][1] = __builtin_amdgcn_wmma_f32_16x16x32_bf16(false, a0, false, b1, (short)0, c[0][1], false, false);
    c[1][0] = __builtin_amdgcn_wmma_f32_16x16x32_bf16(false, a1, false, b0, (short)0, c[1][0], false, false);
    c[1][1] = __builtin_amdgcn_wmma_f32_16x16x32_bf16(false, a1, false, b1, (short)0, c[1][1], false, false);
  }
#pragma unroll
  for (int i = 0; i < 2; ++i)
#pragma unroll
    for (int j = 0; j < 2; ++j) {
      const int n = Nb + j * 16 + l16;
      const float sw = wsc[n], bb = bias[n];
#pragma unroll
      for (int r = 0; r < 8; ++r) {
        const int m = Mb + i * 16 + 8 * lh + r;
        const int b = m >> 10;
        const float val = c[i][j][r] * sw + bb;
        const float g = emb[(size_t)b * NQKV + 2 * DIMC + n];
        outp[(size_t)m * DIMC + n] = hs[(size_t)m * DIMC + n] + g * val;
      }
    }
}

extern "C" void kernel_launch(void* const* d_in, const int* in_sizes, int n_in,
                              void* d_out, int out_size, void* d_ws, size_t ws_size,
                              hipStream_t stream) {
  const float* hs       = (const float*)d_in[0];
  const float* temb     = (const float*)d_in[1];
  const float* rope_cos = (const float*)d_in[2];
  const float* rope_sin = (const float*)d_in[3];
  const float* adaln_w  = (const float*)d_in[4];
  const float* adaln_b  = (const float*)d_in[5];
  const float* wq       = (const float*)d_in[6];
  const float* bq       = (const float*)d_in[7];
  const float* wk       = (const float*)d_in[8];
  const float* bk       = (const float*)d_in[9];
  const float* wv       = (const float*)d_in[10];
  const float* bv       = (const float*)d_in[11];
  const float* rmsq_w   = (const float*)d_in[12];
  const float* rmsk_w   = (const float*)d_in[13];
  const float* wmlp     = (const float*)d_in[14];
  const float* bmlp     = (const float*)d_in[15];
  const float* wout     = (const float*)d_in[16];
  const float* bout     = (const float*)d_in[17];

  char* ws = (char*)d_ws;
  float*       emb   = (float*)(ws + OFF_EMB);
  float*       xs    = (float*)(ws + OFF_XSCALE);
  float*       w8s   = (float*)(ws + OFF_W8S);
  float*       wmlps = (float*)(ws + OFF_WMLPS);
  float*       wouts = (float*)(ws + OFF_WOUTS);
  __bf16*      nx    = (__bf16*)(ws + OFF_NX);
  signed char* xq    = (signed char*)(ws + OFF_XQ);
  signed char* w8    = (signed char*)(ws + OFF_W8);
  __bf16*      wmlpb = (__bf16*)(ws + OFF_WMLP);
  __bf16*      woutb = (__bf16*)(ws + OFF_WOUT);
  float*       qkv   = (float*)(ws + OFF_QKV);
  _Float16*    qh    = (_Float16*)(ws + OFF_QH);
  _Float16*    kh    = (_Float16*)(ws + OFF_KH);
  _Float16*    vh    = (_Float16*)(ws + OFF_VH);
  float*       attn  = (float*)(ws + OFF_ATTN);
  __bf16*      cat   = (__bf16*)(ws + OFF_CAT);
  float*       outp  = (float*)d_out;

  k_wq8  <<<NQKV, 256, 0, stream>>>(wq, wk, wv, w8, w8s);
  k_wqbf <<<MLPD, 256, 0, stream>>>(wmlp, wmlpb, wmlps, DIMC);
  k_wqbf <<<DIMC, 256, 0, stream>>>(wout, woutb, wouts, CATD);
  k_adaln<<<NQKV, 256, 0, stream>>>(temb, adaln_w, adaln_b, emb);
  k_ln   <<<MTOK, 256, 0, stream>>>(hs, emb, nx, xq, xs);
  k_gemm_qkv<<<dim3(MTOK / 128, NQKV / 64), 256, 0, stream>>>(xq, xs, w8, w8s, bq, bk, bv, qkv);
  k_gemm_mlp<<<dim3(MTOK / 128, MLPD / 64), 256, 0, stream>>>(nx, wmlpb, wmlps, bmlp, cat);
  k_qkvhead <<<MTOK * HEADS, 128, 0, stream>>>(qkv, rmsq_w, rmsk_w, rope_cos, rope_sin, qh, kh, vh);
  k_attn    <<<dim3(BATCH * HEADS, SEQ / 128), 256, 0, stream>>>(qh, kh, vh, attn);
  k_attnq   <<<MTOK, 256, 0, stream>>>(attn, cat);
  k_gemm_out<<<dim3(MTOK / 128, DIMC / 64), 256, 0, stream>>>(cat, woutb, wouts, bout, hs, emb, outp);
}
